// GCNModel_58798102282556
// MI455X (gfx1250) — matrix operand order repaired and checked by reading, not compiled, not run on hardware
//
#include <hip/hip_runtime.h>
#include <stdint.h>

#define IN_CH 512
#define HID   64

typedef __attribute__((ext_vector_type(16))) __bf16 v16bf;
typedef __attribute__((ext_vector_type(8)))  float  v8f;
typedef __attribute__((ext_vector_type(4)))  float  v4f;   // native ext-vector for nontemporal loads

union BF16Frag {
    v16bf v;
    unsigned int u[8];
    uint4 q[2];
};

// round-to-nearest-even f32 -> bf16, packed pair into one dword
__device__ __forceinline__ unsigned bf16pack(float lo, float hi) {
    unsigned a = __float_as_uint(lo);
    unsigned b = __float_as_uint(hi);
    a += 0x7FFFu + ((a >> 16) & 1u);
    b += 0x7FFFu + ((b >> 16) & 1u);
    return (a >> 16) | (b & 0xFFFF0000u);
}

__device__ __forceinline__ unsigned short bf16one(float f) {
    unsigned a = __float_as_uint(f);
    a += 0x7FFFu + ((a >> 16) & 1u);
    return (unsigned short)(a >> 16);
}

// load+pack one 16-bit A fragment (16x32 layout) for row pointer `xrow`
__device__ __forceinline__ void load_packA(const float* xrow, int k0, BF16Frag& A) {
    v4f a0 = __builtin_nontemporal_load((const v4f*)(xrow + k0));
    v4f a1 = __builtin_nontemporal_load((const v4f*)(xrow + k0 + 4));
    v4f a2 = __builtin_nontemporal_load((const v4f*)(xrow + k0 + 16));
    v4f a3 = __builtin_nontemporal_load((const v4f*)(xrow + k0 + 20));
    A.u[0] = bf16pack(a0.x, a0.y);
    A.u[1] = bf16pack(a0.z, a0.w);
    A.u[2] = bf16pack(a1.x, a1.y);
    A.u[3] = bf16pack(a1.z, a1.w);
    A.u[4] = bf16pack(a2.x, a2.y);
    A.u[5] = bf16pack(a2.z, a2.w);
    A.u[6] = bf16pack(a3.x, a3.y);
    A.u[7] = bf16pack(a3.z, a3.w);
}

// ---------------- degree / normalization ----------------

__global__ void k_fill1(float* __restrict__ deg, int n) {
    int i = blockIdx.x * blockDim.x + threadIdx.x;
    if (i < n) deg[i] = 1.0f;            // self-loop contributes 1
}

__global__ void k_deg(const int* __restrict__ ei, float* __restrict__ deg, int ne) {
    int e = blockIdx.x * blockDim.x + threadIdx.x;
    if (e < ne) atomicAdd(&deg[ei[ne + e]], 1.0f);   // dst row of edge_index
}

__global__ void k_rsqrt(float* __restrict__ d, int n) {
    int i = blockIdx.x * blockDim.x + threadIdx.x;
    if (i < n) d[i] = rsqrtf(d[i]);
}

// ---------------- W1 -> bf16, transposed [n][k] ----------------

__global__ void k_w1t(const float* __restrict__ W1, unsigned short* __restrict__ w1t, int total) {
    int tid = blockIdx.x * blockDim.x + threadIdx.x;
    if (tid >= total) return;
    int n = tid >> 9;        // 0..63
    int k = tid & 511;       // 0..511
    w1t[tid] = bf16one(W1[(size_t)k * HID + n]);
}

// ---------------- GEMM1: h = x @ W1 via v_wmma_f32_16x16x32_bf16 ----------------
// block = 256 threads = 8 waves; wave -> 32 rows (2 M-tiles), full N=64 (4 n-tiles)
// Per k-step: 4 B fragments loaded once, 8 WMMAs issued (2 M x 4 N).

__global__ void __launch_bounds__(256)
k_gemm1(const float* __restrict__ x, const unsigned short* __restrict__ w1t,
        float* __restrict__ h, int n_nodes) {
    const int wave = threadIdx.x >> 5;
    const int lane = threadIdx.x & 31;
    const int row0 = blockIdx.x * 256 + wave * 32;
    if (row0 >= n_nodes) return;                     // wave-uniform exit (100000 % 32 == 0)

    const int m    = lane & 15;
    const int half = lane >> 4;                      // 0 or 1

    int arow0 = row0 + m;
    int arow1 = row0 + 16 + m;
    if (arow0 >= n_nodes) arow0 = n_nodes - 1;       // defensive clamps
    if (arow1 >= n_nodes) arow1 = n_nodes - 1;
    const float* xrow0 = x + (size_t)arow0 * IN_CH;
    const float* xrow1 = x + (size_t)arow1 * IN_CH;

    v8f acc0[4] = {};
    v8f acc1[4] = {};

    #pragma unroll 4
    for (int ks = 0; ks < 16; ++ks) {
        const int k0 = ks * 32 + half * 8;
        const int kb = ks * 32 + half * 8;

        if (ks < 15) {
            __builtin_prefetch(xrow0 + k0 + 32, 0, 1);   // global_prefetch next k-step
            __builtin_prefetch(xrow1 + k0 + 32, 0, 1);
        }

        // hoist all 4 B fragments (one clause of 8 b128 loads, L1/L2-resident)
        BF16Frag B[4];
        #pragma unroll
        for (int nt = 0; nt < 4; ++nt) {
            const uint4* bp = (const uint4*)(w1t + (size_t)(nt * 16 + m) * IN_CH + kb);
            B[nt].q[0] = bp[0];
            B[nt].q[1] = bp[2];
        }

        // A fragments for both M-tiles; packing VALU overlaps the B loads
        BF16Frag A0, A1;
        load_packA(xrow0, k0, A0);
        load_packA(xrow1, k0, A1);

        #pragma unroll
        for (int nt = 0; nt < 4; ++nt)
            acc0[nt] = __builtin_amdgcn_wmma_f32_16x16x32_bf16(
                false, A0.v, false, B[nt].v, (short)0, acc0[nt], false, false);
        #pragma unroll
        for (int nt = 0; nt < 4; ++nt)
            acc1[nt] = __builtin_amdgcn_wmma_f32_16x16x32_bf16(
                false, A1.v, false, B[nt].v, (short)0, acc1[nt], false, false);
    }

    // C/D layout: col = lane%16, row = r + half*8
    #pragma unroll
    for (int nt = 0; nt < 4; ++nt) {
        #pragma unroll
        for (int r = 0; r < 8; ++r) {
            int mrow0 = row0 + r + half * 8;
            int mrow1 = row0 + 16 + r + half * 8;
            if (mrow0 < n_nodes)
                h[(size_t)mrow0 * HID + nt * 16 + m] = acc0[nt][r];
            if (mrow1 < n_nodes)
                h[(size_t)mrow1 * HID + nt * 16 + m] = acc1[nt][r];
        }
    }
}

// ---------------- aggregation layer 1 ----------------

__global__ void k_selfinit(const float* __restrict__ h, const float* __restrict__ dinv,
                           float* __restrict__ agg, int nn) {
    int tid = blockIdx.x * blockDim.x + threadIdx.x;
    if (tid >= nn * 16) return;
    int node = tid >> 4;
    int c    = (tid & 15) << 2;
    float s  = dinv[node];
    s *= s;
    v4f v = *(const v4f*)(h + (size_t)node * HID + c);
    v4f o = v * s;
    *(v4f*)(agg + (size_t)node * HID + c) = o;
}

__global__ void k_agg1(const int* __restrict__ ei, const float* __restrict__ dinv,
                       const float* __restrict__ h, float* __restrict__ agg, int ne) {
    int tid = blockIdx.x * blockDim.x + threadIdx.x;
    if (tid >= ne * 16) return;
    int e = tid >> 4;
    int c = (tid & 15) << 2;
    int s = ei[e];
    int d = ei[ne + e];
    float nrm = dinv[s] * dinv[d];
    v4f v = *(const v4f*)(h + (size_t)s * HID + c);
    float* o = agg + (size_t)d * HID + c;
    atomicAdd(o + 0, nrm * v.x);
    atomicAdd(o + 1, nrm * v.y);
    atomicAdd(o + 2, nrm * v.z);
    atomicAdd(o + 3, nrm * v.w);
}

// ---------------- layer 2: z = relu(agg+b1) . W2 ; init out with self-loop ----------------

__global__ void k_layer2(const float* __restrict__ agg, const float* __restrict__ b1,
                         const float* __restrict__ W2, const float* __restrict__ dinv,
                         float* __restrict__ z, float* __restrict__ out, int nn) {
    int i = blockIdx.x * blockDim.x + threadIdx.x;
    if (i >= nn) return;
    const float* row = agg + (size_t)i * HID;
    float acc = 0.0f;
    #pragma unroll
    for (int c = 0; c < HID; ++c) {
        float t = row[c] + b1[c];
        t = t > 0.0f ? t : 0.0f;
        acc += t * W2[c];
    }
    z[i] = acc;
    float s = dinv[i];
    out[i] = s * s * acc;                 // self-loop term of conv2
}

__global__ void k_agg2(const int* __restrict__ ei, const float* __restrict__ dinv,
                       const float* __restrict__ z, float* __restrict__ out, int ne) {
    int e = blockIdx.x * blockDim.x + threadIdx.x;
    if (e >= ne) return;
    int s = ei[e];
    int d = ei[ne + e];
    atomicAdd(&out[d], dinv[s] * dinv[d] * z[s]);
}

__global__ void k_sigmoid(float* __restrict__ out, const float* __restrict__ b2, int n) {
    int i = blockIdx.x * blockDim.x + threadIdx.x;
    if (i < n) out[i] = 1.0f / (1.0f + expf(-(out[i] + b2[0])));
}

// ---------------- launcher ----------------

extern "C" void kernel_launch(void* const* d_in, const int* in_sizes, int n_in,
                              void* d_out, int out_size, void* d_ws, size_t ws_size,
                              hipStream_t stream) {
    const float* x  = (const float*)d_in[0];
    const int*   ei = (const int*)d_in[1];
    const float* W1 = (const float*)d_in[2];
    const float* b1 = (const float*)d_in[3];
    const float* W2 = (const float*)d_in[4];
    const float* b2 = (const float*)d_in[5];
    float* out = (float*)d_out;

    const int NN = in_sizes[0] / IN_CH;     // 100000
    const int NE = in_sizes[1] / 2;         // 1600000
    if (NN <= 0) return;

    // workspace carve-out (256B aligned slices)
    uintptr_t p = (uintptr_t)d_ws;
    auto take = [&](size_t bytes) -> uintptr_t {
        uintptr_t r = p;
        p = (p + bytes + 255) & ~(uintptr_t)255;
        return r;
    };
    float*          dinv = (float*)take((size_t)NN * 4);
    float*          z    = (float*)take((size_t)NN * 4);
    unsigned short* w1t  = (unsigned short*)take((size_t)IN_CH * HID * 2);
    float*          h    = (float*)take((size_t)NN * HID * 4);
    float*          agg1 = (float*)take((size_t)NN * HID * 4);
    (void)ws_size; (void)n_in; (void)out_size;

    const int T = 256;

    k_fill1 <<<(NN + T - 1) / T, T, 0, stream>>>(dinv, NN);
    k_deg   <<<(NE + T - 1) / T, T, 0, stream>>>(ei, dinv, NE);
    k_rsqrt <<<(NN + T - 1) / T, T, 0, stream>>>(dinv, NN);

    k_w1t   <<<(IN_CH * HID + T - 1) / T, T, 0, stream>>>(W1, w1t, IN_CH * HID);

    k_gemm1 <<<(NN + 255) / 256, 256, 0, stream>>>(x, w1t, h, NN);

    k_selfinit <<<((NN * 16) + T - 1) / T, T, 0, stream>>>(h, dinv, agg1, NN);
    k_agg1     <<<((NE * 16) + T - 1) / T, T, 0, stream>>>(ei, dinv, h, agg1, NE);

    k_layer2 <<<(NN + T - 1) / T, T, 0, stream>>>(agg1, b1, W2, dinv, z, out, NN);
    k_agg2   <<<(NE + T - 1) / T, T, 0, stream>>>(ei, dinv, z, out, NE);
    k_sigmoid<<<(NN + T - 1) / T, T, 0, stream>>>(out, b2, NN);
}
